// CausalSelfAttention_48163763257346
// MI455X (gfx1250) — matrix operand order repaired and checked by reading, not compiled, not run on hardware
//
#include <hip/hip_runtime.h>
#include <hip/hip_bf16.h>

// ---------------------------------------------------------------------------
// CDNA5 (gfx1250) bf16 WMMA implementation of sliding-window GQA attention.
// All GEMM-like work (QKV proj, Q@K^T, P@V, out proj) runs on
// v_wmma_f32_16x16x32_bf16. wave32 throughout.
// GEMM uses 4x4 WMMA register blocking (64x64/wave): 16 WMMAs per 16 b128
// loads per K-step, so the WMMA pipe (the MI455X ceiling for this
// compute-bound workload) is not vmem-issue limited.
// ---------------------------------------------------------------------------

typedef __bf16 bf16;
typedef bf16 bf16x8 __attribute__((ext_vector_type(8)));
typedef bf16 v16bf  __attribute__((ext_vector_type(16)));
typedef float v8f   __attribute__((ext_vector_type(8)));

#define D_MODEL   2048
#define N_HEADS   16
#define N_KV      4
#define HEAD_DIM  128
#define KV_DIM    512
#define QKV_N     3072          // D_MODEL + 2*KV_DIM
#define T_SEQ     2048
#define BATCH     2
#define WINDOW    512
#define SM_SCALE  0.08838834764831845f   // 1/sqrt(128)

union ABf { bf16x8 h[2]; v16bf v; };

// A-matrix 16x32 bf16 fragment (contiguous k order, not ISA table order): lane = half*16+m, row M=lane%16,
// elem j -> K = half*16 + j (matches load_b_frag).
__device__ inline v16bf load_a_frag(const bf16* base, int lda, int row, int k0, int half) {
    const bf16* p = base + (size_t)row * lda + k0 + half * 16;
    ABf u;
    u.h[0] = *(const bf16x8*)p;
    u.h[1] = *(const bf16x8*)(p + 8);
    return u.v;
}

// B-matrix 32x16 bf16 fragment: N = lane%16, elem j -> K = half*16 + j.
// Requires B stored K-contiguous per column (i.e. B^T row-major, ld = K).
__device__ inline v16bf load_b_frag(const bf16* base, int ldb, int col, int k0, int half) {
    const bf16* p = base + (size_t)col * ldb + k0 + half * 16;
    return *(const v16bf*)p;
}

// ---------------------------------------------------------------------------
// Elementwise conversions
// ---------------------------------------------------------------------------
__global__ void cvt_f32_bf16(const float* __restrict__ in, bf16* __restrict__ out, int n) {
    int i = blockIdx.x * 256 + threadIdx.x;
    if (i < n) out[i] = (bf16)in[i];
}

// in: K x N row-major f32;  out: N x K row-major bf16 (transposed)
__global__ void cvt_transpose(const float* __restrict__ in, bf16* __restrict__ out, int K, int N) {
    int idx = blockIdx.x * 256 + threadIdx.x;
    if (idx >= K * N) return;
    int n = idx / K;
    int k = idx - n * K;
    out[(size_t)n * K + k] = (bf16)in[(size_t)k * N + n];
}

// ---------------------------------------------------------------------------
// bf16 GEMM: C(MxN, f32) = A(MxK bf16 row-major) @ Bt(NxK bf16)^T
// One wave computes a 64x64 tile (4x4 WMMA register blocking).
// ---------------------------------------------------------------------------
__global__ __launch_bounds__(128) void gemm_bf16_f32(
    const bf16* __restrict__ A, const bf16* __restrict__ Bt, float* __restrict__ C,
    int M, int N, int K)
{
    const int wave = threadIdx.x >> 5;
    const int lane = threadIdx.x & 31;
    const int half = lane >> 4;
    const int lid  = lane & 15;

    const int tilesN = N >> 6;
    const int wid = blockIdx.x * 4 + wave;
    const int tm = (wid / tilesN) << 6;
    const int tn = (wid % tilesN) << 6;
    if (tm >= M) return;

    v8f acc[4][4] = {};
    for (int k0 = 0; k0 < K; k0 += 32) {
        v16bf a[4], b[4];
#pragma unroll
        for (int mi = 0; mi < 4; ++mi)
            a[mi] = load_a_frag(A, K, tm + mi * 16 + lid, k0, half);
#pragma unroll
        for (int ni = 0; ni < 4; ++ni)
            b[ni] = load_b_frag(Bt, K, tn + ni * 16 + lid, k0, half);

        if (k0 + 128 < K) {   // pull next stripes toward the WGP (global_prefetch)
            __builtin_prefetch((const void*)(A + (size_t)(tm + lid) * K + k0 + 128), 0, 3);
            __builtin_prefetch((const void*)(Bt + (size_t)(tn + lid) * K + k0 + 128), 0, 3);
        }

#pragma unroll
        for (int mi = 0; mi < 4; ++mi)
#pragma unroll
            for (int ni = 0; ni < 4; ++ni)
                acc[mi][ni] = __builtin_amdgcn_wmma_f32_16x16x32_bf16(
                    false, a[mi], false, b[ni], (short)0, acc[mi][ni], false, false);
    }

    // C/D layout: lane -> N = lane%16; VGPR r -> M = r + 8*half
#pragma unroll
    for (int mi = 0; mi < 4; ++mi)
#pragma unroll
        for (int ni = 0; ni < 4; ++ni)
#pragma unroll
            for (int r = 0; r < 8; ++r) {
                int row = tm + mi * 16 + r + half * 8;
                int col = tn + ni * 16 + lid;
                C[(size_t)row * N + col] = acc[mi][ni][r];
            }
}

// ---------------------------------------------------------------------------
// RoPE + repack: qkv f32 [b][t][3072] ->
//   qb [b][h][t][d] bf16 (rope), kb [b][kvh][t][d] bf16 (rope),
//   vT [b][kvh][d][t] bf16 (transposed, no rope)
// ---------------------------------------------------------------------------
__global__ __launch_bounds__(256) void rope_pack(
    const float* __restrict__ qkv, bf16* __restrict__ qb,
    bf16* __restrict__ kb, bf16* __restrict__ vT)
{
    const int bt = blockIdx.x;
    const int b = bt / T_SEQ, t = bt - b * T_SEQ;
    const float* row = qkv + (size_t)bt * QKV_N;

    for (int i = threadIdx.x; i < D_MODEL; i += 256) {
        int h = i >> 7, d = i & 127;
        float inv = __powf(10000.0f, -(float)(d & 63) * (1.0f / 64.0f));
        float ang = (float)t * inv;
        float c = __cosf(ang), s = __sinf(ang);
        float xr = (d < 64) ? -row[i + 64] : row[i - 64];
        float o = row[i] * c + xr * s;
        qb[((size_t)(b * N_HEADS + h) * T_SEQ + t) * HEAD_DIM + d] = (bf16)o;
    }
    for (int i = threadIdx.x; i < KV_DIM; i += 256) {
        int h = i >> 7, d = i & 127;
        float inv = __powf(10000.0f, -(float)(d & 63) * (1.0f / 64.0f));
        float ang = (float)t * inv;
        float c = __cosf(ang), s = __sinf(ang);
        const float* krow = row + D_MODEL;
        float xr = (d < 64) ? -krow[i + 64] : krow[i - 64];
        float ko = krow[i] * c + xr * s;
        kb[((size_t)(b * N_KV + h) * T_SEQ + t) * HEAD_DIM + d] = (bf16)ko;
        float vv = row[D_MODEL + KV_DIM + i];
        vT[((size_t)(b * N_KV + h) * HEAD_DIM + d) * T_SEQ + t] = (bf16)vv;
    }
}

// ---------------------------------------------------------------------------
// Sliding-window flash attention. One wave per (b, h, 16-query block).
// S = Q@K^T via WMMA; online softmax (half-wave shfl reductions);
// P routed D-layout -> LDS -> A-layout; Y += P@V via WMMA.
// ---------------------------------------------------------------------------
__global__ __launch_bounds__(128) void attn_kernel(
    const bf16* __restrict__ qb, const bf16* __restrict__ kb,
    const bf16* __restrict__ vT, bf16* __restrict__ yb)
{
    __shared__ bf16 Plds[4][16][32];

    const int wave = threadIdx.x >> 5;
    const int lane = threadIdx.x & 31;
    const int half = lane >> 4;
    const int lid  = lane & 15;

    const int qblocks = T_SEQ / 16;
    const int wid = blockIdx.x * 4 + wave;
    const int qbIdx = wid % qblocks;
    const int bh = wid / qblocks;
    const int h = bh % N_HEADS;
    const int b = bh / N_HEADS;
    const int kvh = h / (N_HEADS / N_KV);

    const bf16* qbase = qb + (size_t)(b * N_HEADS + h) * T_SEQ * HEAD_DIM;
    const bf16* kbase = kb + (size_t)(b * N_KV + kvh) * T_SEQ * HEAD_DIM;
    const bf16* vbase = vT + (size_t)(b * N_KV + kvh) * HEAD_DIM * T_SEQ;

    const int qmin = qbIdx * 16, qmax = qmin + 15;

    // Q fragments for all of D=128 (4 K-steps of 32), held for the whole loop.
    v16bf aq[4];
#pragma unroll
    for (int i = 0; i < 4; ++i)
        aq[i] = load_a_frag(qbase, HEAD_DIM, qmin + lid, i * 32, half);

    v8f accY[8] = {};                 // 16 x 128 f32 output tile
    float run_max[8], run_sum[8];
#pragma unroll
    for (int r = 0; r < 8; ++r) { run_max[r] = -1e30f; run_sum[r] = 0.0f; }

    int kstart = qmin - (WINDOW - 1);
    if (kstart < 0) kstart = 0;
    kstart &= ~31;                    // 32-key chunks

    for (int kc = kstart; kc <= qmax; kc += 32) {
        // --- scores for 32 keys: two 16x16 WMMA tiles, K=128 in 4 steps ---
        v8f s0 = {}, s1 = {};
#pragma unroll
        for (int i = 0; i < 4; ++i) {
            v16bf bk0 = load_b_frag(kbase, HEAD_DIM, kc + lid,      i * 32, half);
            v16bf bk1 = load_b_frag(kbase, HEAD_DIM, kc + 16 + lid, i * 32, half);
            s0 = __builtin_amdgcn_wmma_f32_16x16x32_bf16(false, aq[i], false, bk0, (short)0, s0, false, false);
            s1 = __builtin_amdgcn_wmma_f32_16x16x32_bf16(false, aq[i], false, bk1, (short)0, s1, false, false);
        }

        // --- mask + online softmax (rows live in 16-lane halves) ---
        float p0[8], p1[8];
#pragma unroll
        for (int r = 0; r < 8; ++r) {
            const int trow = qmin + r + half * 8;
            const int tk0 = kc + lid, tk1 = kc + 16 + lid;
            float sv0 = s0[r] * SM_SCALE;
            float sv1 = s1[r] * SM_SCALE;
            sv0 = (tk0 <= trow && tk0 >= trow - (WINDOW - 1)) ? sv0 : -1e30f;
            sv1 = (tk1 <= trow && tk1 >= trow - (WINDOW - 1)) ? sv1 : -1e30f;
            float m = fmaxf(sv0, sv1);
            for (int off = 1; off < 16; off <<= 1)
                m = fmaxf(m, __shfl_xor(m, off, 32));
            float nm = fmaxf(run_max[r], m);
            float rescale = __expf(run_max[r] - nm);
            float e0 = (sv0 > -1e29f) ? __expf(sv0 - nm) : 0.0f;
            float e1 = (sv1 > -1e29f) ? __expf(sv1 - nm) : 0.0f;
            p0[r] = e0; p1[r] = e1;
            float lsum = e0 + e1;
            for (int off = 1; off < 16; off <<= 1)
                lsum += __shfl_xor(lsum, off, 32);
            run_sum[r] = run_sum[r] * rescale + lsum;
            run_max[r] = nm;
#pragma unroll
            for (int nt = 0; nt < 8; ++nt)
                accY[nt][r] *= rescale;
        }

        // --- D-layout P -> LDS -> A-layout fragment ---
#pragma unroll
        for (int r = 0; r < 8; ++r) {
            Plds[wave][r + half * 8][lid]      = (bf16)p0[r];
            Plds[wave][r + half * 8][16 + lid] = (bf16)p1[r];
        }
        ABf ua;
        ua.h[0] = *(const bf16x8*)&Plds[wave][lid][half * 16];
        ua.h[1] = *(const bf16x8*)&Plds[wave][lid][half * 16 + 8];
        v16bf pa = ua.v;

        // --- Y += P(16x32) @ V(32x128): 8 WMMAs over d-tiles ---
#pragma unroll
        for (int nt = 0; nt < 8; ++nt) {
            v16bf bv = load_b_frag(vbase, T_SEQ, nt * 16 + lid, kc, half);
            accY[nt] = __builtin_amdgcn_wmma_f32_16x16x32_bf16(false, pa, false, bv, (short)0, accY[nt], false, false);
        }
    }

    // normalize + store y [b][t][d_model] bf16
    const size_t ybase = ((size_t)b * T_SEQ) * D_MODEL + (size_t)h * HEAD_DIM;
#pragma unroll
    for (int r = 0; r < 8; ++r) {
        float inv = 1.0f / run_sum[r];
        int trow = qmin + r + half * 8;
#pragma unroll
        for (int nt = 0; nt < 8; ++nt)
            yb[ybase + (size_t)trow * D_MODEL + nt * 16 + lid] = (bf16)(accY[nt][r] * inv);
    }
}

// ---------------------------------------------------------------------------
extern "C" void kernel_launch(void* const* d_in, const int* in_sizes, int n_in,
                              void* d_out, int out_size, void* d_ws, size_t ws_size,
                              hipStream_t stream) {
    const float* x    = (const float*)d_in[0];
    const float* Wqkv = (const float*)d_in[1];
    const float* Wout = (const float*)d_in[2];
    float* out = (float*)d_out;

    char* ws = (char*)d_ws;
    size_t off = 0;
    auto wsAlloc = [&](size_t bytes) -> void* {
        void* p = ws + off;
        off += (bytes + 255) & ~(size_t)255;
        return p;
    };
    bf16*  xbf   = (bf16*) wsAlloc((size_t)BATCH * T_SEQ * D_MODEL * 2);
    bf16*  WqkvT = (bf16*) wsAlloc((size_t)QKV_N * D_MODEL * 2);
    bf16*  WoutT = (bf16*) wsAlloc((size_t)D_MODEL * D_MODEL * 2);
    float* qkv   = (float*)wsAlloc((size_t)BATCH * T_SEQ * QKV_N * 4);
    bf16*  qb    = (bf16*) wsAlloc((size_t)BATCH * N_HEADS * T_SEQ * HEAD_DIM * 2);
    bf16*  kb    = (bf16*) wsAlloc((size_t)BATCH * N_KV * T_SEQ * HEAD_DIM * 2);
    bf16*  vT    = (bf16*) wsAlloc((size_t)BATCH * N_KV * HEAD_DIM * T_SEQ * 2);
    bf16*  yb    = (bf16*) wsAlloc((size_t)BATCH * T_SEQ * D_MODEL * 2);

    const int nx = BATCH * T_SEQ * D_MODEL;
    cvt_f32_bf16<<<(nx + 255) / 256, 256, 0, stream>>>(x, xbf, nx);
    cvt_transpose<<<(D_MODEL * QKV_N + 255) / 256, 256, 0, stream>>>(Wqkv, WqkvT, D_MODEL, QKV_N);
    cvt_transpose<<<(D_MODEL * D_MODEL + 255) / 256, 256, 0, stream>>>(Wout, WoutT, D_MODEL, D_MODEL);

    {   // qkv = x @ Wqkv : 64x64 tiles -> (4096/64)*(3072/64) = 3072 waves
        const int M = BATCH * T_SEQ, N = QKV_N, K = D_MODEL;
        const int waves = (M / 64) * (N / 64);
        gemm_bf16_f32<<<waves / 4, 128, 0, stream>>>(xbf, WqkvT, qkv, M, N, K);
    }

    rope_pack<<<BATCH * T_SEQ, 256, 0, stream>>>(qkv, qb, kb, vT);

    {   // attention: 2*16*128 = 4096 waves
        const int waves = BATCH * N_HEADS * (T_SEQ / 16);
        attn_kernel<<<waves / 4, 128, 0, stream>>>(qb, kb, vT, yb);
    }

    {   // out = y @ Wout : (4096/64)*(2048/64) = 2048 waves
        const int M = BATCH * T_SEQ, N = D_MODEL, K = D_MODEL;
        const int waves = (M / 64) * (N / 64);
        gemm_bf16_f32<<<waves / 4, 128, 0, stream>>>(yb, WoutT, out, M, N, K);
    }
}
